// AttentionSetOfSetLayer_83588653515271
// MI455X (gfx1250) — compile-verified, matrix operand order checked by reading, not run on hardware
//
#include <hip/hip_runtime.h>
#include <hip/hip_bf16.h>

typedef __attribute__((ext_vector_type(16))) __bf16 v16bf;
typedef __attribute__((ext_vector_type(8)))  float  v8f;

__device__ __forceinline__ unsigned short f2bf(float f) {
    __bf16 b = (__bf16)f;
    unsigned short u;
    __builtin_memcpy(&u, &b, 2);
    return u;
}
__device__ __forceinline__ float bf2f(unsigned short u) {
    __bf16 b;
    __builtin_memcpy(&b, &u, 2);
    return (float)b;
}

union FragU  { uint4 q[2]; v16bf v; };
union Pack8U { uint4 q; unsigned short u[8]; };

// ---------------- utility kernels ----------------

__global__ void zero_f32_kernel(float* p, long long n) {
    long long i = (long long)blockIdx.x * blockDim.x + threadIdx.x;
    long long stride = (long long)gridDim.x * blockDim.x;
    for (; i < n; i += stride) p[i] = 0.0f;
}

__global__ void convert_bf16_kernel(const float* __restrict__ src,
                                    unsigned short* __restrict__ dst, int n) {
    int i = blockIdx.x * blockDim.x + threadIdx.x;
    int stride = gridDim.x * blockDim.x;
    for (; i < n; i += stride) dst[i] = f2bf(src[i]);
}

// one block per edge: 256 threads scatter one 256-float row into both node aggs
__global__ void __launch_bounds__(256)
scatter_kernel(const float* __restrict__ ef, const int* __restrict__ cidx,
               const int* __restrict__ pidx, float* cam_agg, float* pt_agg, int E) {
    int e = blockIdx.x;
    if (e >= E) return;
    int t = threadIdx.x;
    float v = ef[(size_t)e * 256 + t];
    int c = cidx[e];
    int p = pidx[e];
    atomicAdd(&cam_agg[(size_t)c * 256 + t], v);
    atomicAdd(&pt_agg[(size_t)p * 256 + t], v);
}

// ---------------- GEMM1: H[M,256](bf16) = A[M,256](f32) @ B.T, B=[256,256](bf16 rows=N) ----
// one wave -> one 16x16 tile; 4 waves/block along M; grid.y = N/16 = 16

__global__ void __launch_bounds__(128)
gemm_agg_kernel(const float* __restrict__ A, const unsigned short* __restrict__ B,
                unsigned short* __restrict__ H, int Mtiles) {
    const int lane = threadIdx.x & 31;
    const int wave = threadIdx.x >> 5;
    const int mt = blockIdx.x * 4 + wave;
    if (mt >= Mtiles) return;                 // wave-uniform
    const int nt  = blockIdx.y;
    const int m15 = lane & 15;
    const int g   = lane >> 4;

    const float*          arow = A + (size_t)(mt * 16 + m15) * 256;
    const unsigned short* brow = B + (size_t)(nt * 16 + m15) * 256;

    v8f acc = {};
#pragma unroll
    for (int kt = 0; kt < 8; ++kt) {
        const int k0 = kt * 32 + g * 8;
        float4 a0 = *(const float4*)(arow + k0);
        float4 a1 = *(const float4*)(arow + k0 + 4);
        float4 a2 = *(const float4*)(arow + k0 + 16);
        float4 a3 = *(const float4*)(arow + k0 + 20);
        v16bf af;
        af[0]=(__bf16)a0.x;  af[1]=(__bf16)a0.y;  af[2]=(__bf16)a0.z;  af[3]=(__bf16)a0.w;
        af[4]=(__bf16)a1.x;  af[5]=(__bf16)a1.y;  af[6]=(__bf16)a1.z;  af[7]=(__bf16)a1.w;
        af[8]=(__bf16)a2.x;  af[9]=(__bf16)a2.y;  af[10]=(__bf16)a2.z; af[11]=(__bf16)a2.w;
        af[12]=(__bf16)a3.x; af[13]=(__bf16)a3.y; af[14]=(__bf16)a3.z; af[15]=(__bf16)a3.w;

        FragU bu;
        bu.q[0] = *(const uint4*)(brow + k0);
        bu.q[1] = *(const uint4*)(brow + k0 + 16);

        acc = __builtin_amdgcn_wmma_f32_16x16x32_bf16(false, af, false, bu.v,
                                                      (short)0, acc, false, false);
    }
#pragma unroll
    for (int r = 0; r < 8; ++r) {
        int m = mt * 16 + g * 8 + r;
        H[(size_t)m * 256 + nt * 16 + m15] = f2bf(acc[r]);
    }
}

// ---------------- attention scores: e[i] = concat_i . W_attn ----------------

__global__ void __launch_bounds__(256)
score_kernel(const unsigned short* __restrict__ hc, const unsigned short* __restrict__ hp,
             const int* __restrict__ cidx, const int* __restrict__ pidx,
             const float* __restrict__ Wa, float* __restrict__ scores, int E) {
    __shared__ float wa[512];
    for (int i = threadIdx.x; i < 512; i += blockDim.x) wa[i] = Wa[i];
    __syncthreads();
    int e = blockIdx.x * blockDim.x + threadIdx.x;
    if (e >= E) return;
    const unsigned short* cr = hc + (size_t)cidx[e] * 256;
    const unsigned short* pr = hp + (size_t)pidx[e] * 256;
    float s = 0.0f;
#pragma unroll 4
    for (int k = 0; k < 256; k += 8) {
        Pack8U c8, p8;
        c8.q = *(const uint4*)(cr + k);
        p8.q = *(const uint4*)(pr + k);
#pragma unroll
        for (int j = 0; j < 8; ++j) {
            s += bf2f(c8.u[j]) * wa[k + j];
            s += bf2f(p8.u[j]) * wa[256 + k + j];
        }
    }
    scores[e] = s;
}

// ---------------- softmax reductions ----------------

__global__ void __launch_bounds__(256)
reduce_max_kernel(const float* __restrict__ scores, float* __restrict__ partials, int E) {
    __shared__ float sm[256];
    float m = -3.4e38f;
    for (int i = blockIdx.x * blockDim.x + threadIdx.x; i < E; i += gridDim.x * blockDim.x)
        m = fmaxf(m, scores[i]);
    sm[threadIdx.x] = m;
    __syncthreads();
    for (int s = 128; s > 0; s >>= 1) {
        if (threadIdx.x < s) sm[threadIdx.x] = fmaxf(sm[threadIdx.x], sm[threadIdx.x + s]);
        __syncthreads();
    }
    if (threadIdx.x == 0) partials[blockIdx.x] = sm[0];
}

__global__ void __launch_bounds__(256)
final_max_kernel(const float* __restrict__ partials, float* __restrict__ scal) {
    __shared__ float sm[256];
    sm[threadIdx.x] = partials[threadIdx.x];
    __syncthreads();
    for (int s = 128; s > 0; s >>= 1) {
        if (threadIdx.x < s) sm[threadIdx.x] = fmaxf(sm[threadIdx.x], sm[threadIdx.x + s]);
        __syncthreads();
    }
    if (threadIdx.x == 0) scal[0] = sm[0];
}

__global__ void __launch_bounds__(256)
reduce_sum_kernel(const float* __restrict__ scores, const float* __restrict__ scal,
                  float* __restrict__ partials, int E) {
    __shared__ float sm[256];
    const float gmax = scal[0];
    float acc = 0.0f;
    for (int i = blockIdx.x * blockDim.x + threadIdx.x; i < E; i += gridDim.x * blockDim.x)
        acc += __expf(scores[i] - gmax);
    sm[threadIdx.x] = acc;
    __syncthreads();
    for (int s = 128; s > 0; s >>= 1) {
        if (threadIdx.x < s) sm[threadIdx.x] += sm[threadIdx.x + s];
        __syncthreads();
    }
    if (threadIdx.x == 0) partials[blockIdx.x] = sm[0];
}

__global__ void __launch_bounds__(256)
final_sum_kernel(const float* __restrict__ partials, float* __restrict__ scal) {
    __shared__ float sm[256];
    sm[threadIdx.x] = partials[threadIdx.x];
    __syncthreads();
    for (int s = 128; s > 0; s >>= 1) {
        if (threadIdx.x < s) sm[threadIdx.x] += sm[threadIdx.x + s];
        __syncthreads();
    }
    if (threadIdx.x == 0) scal[1] = 1.0f / sm[0];
}

// ---------------- final fused GEMM: out = relu((concat @ Wf.T + b) * attn) -------------
// block = 256 threads = 8 waves, each wave owns its own 16-edge tile and loops over all
// 16 N-tiles, so gathered A fragments (16 K-tiles in registers) are loaded exactly once
// per edge. W_final is staged into LDS in two 128-row phases (bank-padded rows), so
// B fragments come from ds_load_b128 and W_final L2 traffic drops ~8x.

#define LROW 520   // padded LDS row stride in bf16 elems (520*2 B; dword stride 260 == 4 mod 64)

__global__ void __launch_bounds__(256)
edge_final_kernel(const unsigned short* __restrict__ hc, const unsigned short* __restrict__ hp,
                  const int* __restrict__ cidx, const int* __restrict__ pidx,
                  const unsigned short* __restrict__ Wf, const float* __restrict__ bfin,
                  const float* __restrict__ scores, const float* __restrict__ scal,
                  float* __restrict__ out, int E, int Etiles) {
    extern __shared__ unsigned short smem[];   // 128 rows x LROW bf16
    const int tid  = threadIdx.x;
    const int lane = tid & 31;
    const int wave = tid >> 5;
    const int m15  = lane & 15;
    const int g    = lane >> 4;
    const int et   = blockIdx.x * 8 + wave;
    const bool valid = (et < Etiles);          // wave-uniform

    // gather A fragments (clamped addresses keep inactive waves' loads legal)
    const int etc = valid ? et : 0;
    int e = etc * 16 + m15;
    if (e >= E) e = E - 1;
    const unsigned short* crow = hc + (size_t)cidx[e] * 256;
    const unsigned short* prow = hp + (size_t)pidx[e] * 256;

    v16bf afr[16];
#pragma unroll
    for (int kt = 0; kt < 8; ++kt) {
        const int k0 = kt * 32 + g * 8;
        FragU uc, up;
        uc.q[0] = *(const uint4*)(crow + k0);
        uc.q[1] = *(const uint4*)(crow + k0 + 16);
        up.q[0] = *(const uint4*)(prow + k0);
        up.q[1] = *(const uint4*)(prow + k0 + 16);
        afr[kt]     = uc.v;   // K 0..255   = cam half of concat
        afr[kt + 8] = up.v;   // K 256..511 = point half
    }

    const float gmax = scal[0];
    const float isum = scal[1];
    float sc[8];
#pragma unroll
    for (int r = 0; r < 8; ++r) {
        int ee = etc * 16 + g * 8 + r;
        int ec = ee < E ? ee : E - 1;
        sc[r] = __expf(scores[ec] - gmax) * isum;
    }

    for (int phase = 0; phase < 2; ++phase) {
        __syncthreads();   // LDS safe to overwrite (uniform point, all threads)
        // stage W_final rows [phase*128, phase*128+128) -> LDS (padded rows)
        for (int i = tid; i < 8192; i += 256) {
            int r = i >> 6;          // local row 0..127
            int c = i & 63;          // 16B chunk 0..63
            *(uint4*)((char*)smem + (size_t)r * (LROW * 2) + c * 16) =
                *(const uint4*)(Wf + (size_t)(phase * 128 + r) * 512 + c * 8);
        }
        __syncthreads();

        if (valid) {
            for (int jn = 0; jn < 8; ++jn) {
                const int nt = phase * 8 + jn;
                const unsigned short* brow = smem + (size_t)(jn * 16 + m15) * LROW;
                v8f acc = {};
#pragma unroll
                for (int kt = 0; kt < 16; ++kt) {
                    const int k0 = kt * 32 + g * 8;
                    FragU bu;
                    bu.q[0] = *(const uint4*)(brow + k0);       // ds_load_b128
                    bu.q[1] = *(const uint4*)(brow + k0 + 16);
                    acc = __builtin_amdgcn_wmma_f32_16x16x32_bf16(false, afr[kt], false, bu.v,
                                                                  (short)0, acc, false, false);
                }
                const int n = nt * 16 + m15;
                const float bias = bfin[n];
#pragma unroll
                for (int r = 0; r < 8; ++r) {
                    int ee = et * 16 + g * 8 + r;
                    if (ee < E) {
                        float v = (acc[r] + bias) * sc[r];
                        out[(size_t)ee * 256 + n] = v > 0.0f ? v : 0.0f;
                    }
                }
            }
        }
    }
}

// ---------------- host launcher ----------------

extern "C" void kernel_launch(void* const* d_in, const int* in_sizes, int n_in,
                              void* d_out, int out_size, void* d_ws, size_t ws_size,
                              hipStream_t stream) {
    const float* edge_features = (const float*)d_in[0];
    const int*   cam_indices   = (const int*)d_in[1];
    const int*   point_indices = (const int*)d_in[2];
    // d_in[3]=num_cams, d_in[4]=num_points (device scalars; sizes fixed by problem)
    const float* W_cam   = (const float*)d_in[5];
    const float* W_point = (const float*)d_in[6];
    const float* W_attn  = (const float*)d_in[7];
    const float* W_final = (const float*)d_in[8];
    const float* b_final = (const float*)d_in[9];
    float* out = (float*)d_out;

    const int E   = in_sizes[1];   // 250000
    const int NCP = 512;           // num_cams=500 padded to multiple of 16
    const int NP  = 100000;        // num_points (multiple of 16)

    char* ws = (char*)d_ws;
    size_t off = 0;
    float*          cam_agg  = (float*)(ws + off);           off += (size_t)NCP * 256 * 4;
    float*          pt_agg   = (float*)(ws + off);           off += (size_t)NP  * 256 * 4;
    unsigned short* h_cam    = (unsigned short*)(ws + off);  off += (size_t)NCP * 256 * 2;
    unsigned short* h_point  = (unsigned short*)(ws + off);  off += (size_t)NP  * 256 * 2;
    unsigned short* Wc_bf    = (unsigned short*)(ws + off);  off += 256 * 256 * 2;
    unsigned short* Wp_bf    = (unsigned short*)(ws + off);  off += 256 * 256 * 2;
    unsigned short* Wf_bf    = (unsigned short*)(ws + off);  off += 256 * 512 * 2;
    float*          scores   = (float*)(ws + off);           off += (size_t)E * 4;
    float*          part_max = (float*)(ws + off);           off += 256 * 4;
    float*          part_sum = (float*)(ws + off);           off += 256 * 4;
    float*          scal     = (float*)(ws + off);           off += 16;
    (void)ws_size;

    // 1) zero aggregation buffers (cam padded rows included)
    long long nzero = (long long)(NCP + NP) * 256;
    zero_f32_kernel<<<4096, 256, 0, stream>>>(cam_agg, nzero);

    // 2) weights -> bf16
    convert_bf16_kernel<<<256, 256, 0, stream>>>(W_cam,   Wc_bf, 256 * 256);
    convert_bf16_kernel<<<256, 256, 0, stream>>>(W_point, Wp_bf, 256 * 256);
    convert_bf16_kernel<<<256, 256, 0, stream>>>(W_final, Wf_bf, 256 * 512);

    // 3) segment sums via f32 global atomics
    scatter_kernel<<<E, 256, 0, stream>>>(edge_features, cam_indices, point_indices,
                                          cam_agg, pt_agg, E);

    // 4) node projections (WMMA bf16, f32 accumulate)
    {
        int mt = NCP / 16;                                   // 32
        dim3 grid((mt + 3) / 4, 16);
        gemm_agg_kernel<<<grid, 128, 0, stream>>>(cam_agg, Wc_bf, h_cam, mt);
    }
    {
        int mt = NP / 16;                                    // 6250
        dim3 grid((mt + 3) / 4, 16);
        gemm_agg_kernel<<<grid, 128, 0, stream>>>(pt_agg, Wp_bf, h_point, mt);
    }

    // 5) attention scores + global softmax reduction
    score_kernel<<<(E + 255) / 256, 256, 0, stream>>>(h_cam, h_point, cam_indices,
                                                      point_indices, W_attn, scores, E);
    reduce_max_kernel<<<256, 256, 0, stream>>>(scores, part_max, E);
    final_max_kernel<<<1, 256, 0, stream>>>(part_max, scal);
    reduce_sum_kernel<<<256, 256, 0, stream>>>(scores, scal, part_sum, E);
    final_sum_kernel<<<1, 256, 0, stream>>>(part_sum, scal);

    // 6) fused gather + GEMM + bias + softmax-scale + relu (LDS-staged W_final)
    {
        const int Etiles = (E + 15) / 16;                    // 15625
        const int blocks = (Etiles + 7) / 8;                 // 1954
        const size_t smemBytes = 128 * LROW * 2;             // 133,120 B
        edge_final_kernel<<<blocks, 256, smemBytes, stream>>>(h_cam, h_point, cam_indices,
                                                              point_indices, Wf_bf, b_final,
                                                              scores, scal, out, E, Etiles);
    }
}
